// Attention_33646773797316
// MI455X (gfx1250) — matrix operand order repaired and checked by reading, not compiled, not run on hardware
//
#include <hip/hip_runtime.h>

typedef _Float16 f16;
typedef f16  v16h __attribute__((ext_vector_type(16)));
typedef f16  v8h  __attribute__((ext_vector_type(8)));
typedef float v8f __attribute__((ext_vector_type(8)));
typedef float v4f __attribute__((ext_vector_type(4)));

#define G_ 64
#define M_ 16
#define F_ 256
#define D_ 512
#define N_ 50000

__device__ __forceinline__ v8f wmma16(v16h a, v16h b, v8f c) {
  return __builtin_amdgcn_wmma_f32_16x16x32_f16(false, a, false, b, (short)0, c,
                                                false, false);
}

// Build an A-operand fragment (16x32 f16, M = lane&15) from a row pointer at the
// current K offset. base = (lane>>4)*8. a[0..7] = K[base..base+7], a[8..15] = K[16+base..].
__device__ __forceinline__ v16h mk_a(const f16* p_lo, const f16* p_hi) {
  union { v16h v; v8h h[2]; } u;
  u.h[0] = *(const v8h*)p_lo;
  u.h[1] = *(const v8h*)p_hi;
  return u.v;
}

// ---------------- prep kernels ----------------
__global__ void k_cvt_f16(const float* __restrict__ in, f16* __restrict__ out, int n) {
  int t = blockIdx.x * blockDim.x + threadIdx.x;
  if (t < n) out[t] = (f16)in[t];
}

// in: [R][C] f32 row-major  ->  out: [C][R] f16 (out[c][r] = in[r][c])
__global__ void k_transpose_f16(const float* __restrict__ in, f16* __restrict__ out,
                                int R, int C) {
  int t = blockIdx.x * blockDim.x + threadIdx.x;
  if (t >= R * C) return;
  int r = t / C, c = t % C;
  out[(size_t)c * R + r] = (f16)in[(size_t)r * C + c];
}

// ---------------- q/k/v projection ----------------
// one wave per (g, which, dtile): out tile = factors[g] (16x256) x W (256 x 16-slice)
__global__ void k_qkv(const f16* __restrict__ factors_h,
                      const f16* __restrict__ WqT, const f16* __restrict__ WkT,
                      const f16* __restrict__ WvT,
                      f16* __restrict__ q_h, f16* __restrict__ k_h,
                      f16* __restrict__ vT_pad) {
  int bid = blockIdx.x;
  int g = bid / 96;
  int rem = bid % 96;
  int which = rem >> 5;      // 0=q 1=k 2=v
  int dtile = rem & 31;      // 0..31

  int lane = threadIdx.x;
  int nl = lane & 15;
  int half = lane >> 4;
  int base = half * 8;

  const f16* WT = (which == 0) ? WqT : (which == 1) ? WkT : WvT;
  const f16* arow = factors_h + (size_t)g * (M_ * F_) + nl * F_;   // A row m = nl
  int dglob = dtile * 16 + nl;                                      // B row n = nl
  const f16* brow = WT + (size_t)dglob * F_;

  v8f c = {};
#pragma unroll
  for (int kk = 0; kk < F_; kk += 32) {
    v16h a = mk_a(arow + kk + base, arow + kk + 16 + base);
    v16h b = mk_a(brow + kk + base, brow + kk + 16 + base);
    c = wmma16(a, b, c);
  }

  if (which < 2) {
    // row-major f16 [g][16][512] ; C frag: m = half*8+r, col = dglob
    f16* out = ((which == 0) ? q_h : k_h) + (size_t)g * (M_ * D_);
#pragma unroll
    for (int r = 0; r < 8; ++r) out[(half * 8 + r) * D_ + dglob] = (f16)c[r];
  } else {
    // vT_pad [g][512][32]: row d, cols 0..15 = v[m][d], cols 16..31 = 0 (K pad)
    f16* row = vT_pad + (size_t)g * (D_ * 32) + (size_t)dglob * 32;
    v8h t, z;
#pragma unroll
    for (int r = 0; r < 8; ++r) { t[r] = (f16)c[r]; z[r] = (f16)0.f; }
    *(v8h*)(row + half * 8) = t;
    *(v8h*)(row + 16 + half * 8) = z;
  }
}

// ---------------- attention + MLP, one wave per group ----------------
template <int IN, int OUT>
__device__ __forceinline__ void mlp_layer(const f16* a_lds /*[16][IN]*/,
                                          const f16* __restrict__ WT /*[OUT][IN]*/,
                                          const float* __restrict__ bias,
                                          f16* out_lds /*[16][OUT]*/, int lane) {
  int nl = lane & 15, half = lane >> 4, base = half * 8;
  const f16* arow = a_lds + nl * IN;
  for (int ot = 0; ot < OUT / 16; ++ot) {
    int og = ot * 16 + nl;
    const f16* brow = WT + (size_t)og * IN;
    v8f c = {};
#pragma unroll
    for (int kk = 0; kk < IN; kk += 32) {
      v16h a = mk_a(arow + kk + base, arow + kk + 16 + base);
      v16h b = mk_a(brow + kk + base, brow + kk + 16 + base);
      c = wmma16(a, b, c);
    }
    float bv = bias[og];
#pragma unroll
    for (int r = 0; r < 8; ++r)
      out_lds[(half * 8 + r) * OUT + og] = (f16)(c[r] + bv);
  }
}

__global__ void __launch_bounds__(32)
k_att(const f16* __restrict__ q_h, const f16* __restrict__ k_h,
      const f16* __restrict__ vT_pad, const int* __restrict__ lengths,
      const f16* __restrict__ W1T, const float* __restrict__ b1,
      const f16* __restrict__ W2T, const float* __restrict__ b2,
      const f16* __restrict__ W3T, const float* __restrict__ b3,
      const float* __restrict__ W4, const float* __restrict__ b4,
      float* __restrict__ w_out) {
  __shared__ f16 attn_lds[16 * 32];
  __shared__ f16 ctx_lds[16 * 512];
  __shared__ f16 h1_lds[16 * 256];
  __shared__ f16 h2_lds[16 * 128];
  __shared__ f16 h3_lds[16 * 64];

  int g = blockIdx.x;
  int lane = threadIdx.x;
  int nl = lane & 15, half = lane >> 4, base = half * 8;
  int len = lengths[g];

  // ---- scores = q k^T  (16x16, K = 512) ----
  const f16* arow = q_h + (size_t)g * (M_ * D_) + nl * D_;   // A row m = nl
  const f16* brow = k_h + (size_t)g * (M_ * D_) + nl * D_;   // B^T row n = nl
  v8f sc = {};
#pragma unroll
  for (int kk = 0; kk < D_; kk += 32) {
    v16h a = mk_a(arow + kk + base, arow + kk + 16 + base);
    v16h b = mk_a(brow + kk + base, brow + kk + 16 + base);
    sc = wmma16(a, b, sc);
  }

  // ---- masked softmax over n (columns), rows live across 16-lane halves ----
  bool valid = nl < len;
#pragma unroll
  for (int r = 0; r < 8; ++r) {
    float x = valid ? (float)sc[r] : -1e30f;
    float mx = x;
    mx = fmaxf(mx, __shfl_xor(mx, 1, 16));
    mx = fmaxf(mx, __shfl_xor(mx, 2, 16));
    mx = fmaxf(mx, __shfl_xor(mx, 4, 16));
    mx = fmaxf(mx, __shfl_xor(mx, 8, 16));
    float e = __expf(x - mx);
    float s = e;
    s += __shfl_xor(s, 1, 16);
    s += __shfl_xor(s, 2, 16);
    s += __shfl_xor(s, 4, 16);
    s += __shfl_xor(s, 8, 16);
    float av = e / s;
    int m = half * 8 + r;
    attn_lds[m * 32 + nl] = (f16)av;          // real K = 0..15
    attn_lds[m * 32 + 16 + nl] = (f16)0.f;    // zero pad K = 16..31
  }
  __syncthreads();

  // A fragment of attn (16x32, padded)
  v16h a_attn = mk_a(&attn_lds[nl * 32 + base], &attn_lds[nl * 32 + 16 + base]);

  // ---- ctx = attn x v  (16x512) ----
  for (int dt = 0; dt < 32; ++dt) {
    int dg = dt * 16 + nl;
    v16h b = mk_a(vT_pad + (size_t)g * (D_ * 32) + (size_t)dg * 32 + base, vT_pad + (size_t)g * (D_ * 32) + (size_t)dg * 32 + 16 + base);
    v8f c = {};
    c = wmma16(a_attn, b, c);
#pragma unroll
    for (int r = 0; r < 8; ++r) ctx_lds[(half * 8 + r) * D_ + dg] = (f16)c[r];
  }
  __syncthreads();

  // ---- MLP ----
  mlp_layer<512, 256>(ctx_lds, W1T, b1, h1_lds, lane);
  __syncthreads();
  mlp_layer<256, 128>(h1_lds, W2T, b2, h2_lds, lane);
  __syncthreads();
  mlp_layer<128, 64>(h2_lds, W3T, b3, h3_lds, lane);
  __syncthreads();

  // ---- final projection to scalar + mask ----
  if (lane < 16) {
    float s = b4[0];
#pragma unroll
    for (int j = 0; j < 64; ++j) s += (float)h3_lds[lane * 64 + j] * W4[j];
    w_out[g * 16 + lane] = (lane < len) ? s : 0.f;
  }
}

// ---------------- memory-bound weighted reduction over raw ----------------
// raw is 204.8 MB (> 192 MB L2) and read exactly once; out is write-once.
// Use non-temporal cache policy (TH=NT) for both streams so they don't evict
// the small reusable working set from L2.
__global__ void __launch_bounds__(256)
k_final(const float* __restrict__ raw, const float* __restrict__ w,
        float* __restrict__ out) {
  __shared__ float wsh[G_ * M_];
  int tid = threadIdx.x;
#pragma unroll
  for (int i = tid; i < G_ * M_; i += 256) wsh[i] = w[i];
  __syncthreads();

  int t = blockIdx.x * 256 + tid;          // t = n*64 + g, t < N*G
  int g = t & 63;
  const v4f* rp = (const v4f*)raw + (size_t)t * 4;
  const float* wr = &wsh[g * 16];
  float s = 0.f;
#pragma unroll
  for (int i = 0; i < 4; ++i) {
    v4f rv = __builtin_nontemporal_load(rp + i);
    s += rv.x * wr[i * 4 + 0] + rv.y * wr[i * 4 + 1] +
         rv.z * wr[i * 4 + 2] + rv.w * wr[i * 4 + 3];
  }
  __builtin_nontemporal_store(s, &out[t]);
}

// ---------------- host ----------------
extern "C" void kernel_launch(void* const* d_in, const int* in_sizes, int n_in,
                              void* d_out, int out_size, void* d_ws, size_t ws_size,
                              hipStream_t stream) {
  const float* raw     = (const float*)d_in[0];
  const float* factors = (const float*)d_in[1];
  const int*   lengths = (const int*)d_in[2];
  const float* Wq = (const float*)d_in[3];
  const float* Wk = (const float*)d_in[4];
  const float* Wv = (const float*)d_in[5];
  const float* W1 = (const float*)d_in[6];
  const float* b1 = (const float*)d_in[7];
  const float* W2 = (const float*)d_in[8];
  const float* b2 = (const float*)d_in[9];
  const float* W3 = (const float*)d_in[10];
  const float* b3 = (const float*)d_in[11];
  const float* W4 = (const float*)d_in[12];
  const float* b4 = (const float*)d_in[13];
  float* out = (float*)d_out;

  // workspace layout (all sizes 256B-aligned)
  char* ws = (char*)d_ws;
  size_t off = 0;
  auto alloc = [&](size_t bytes) { size_t r = off; off += (bytes + 255) & ~(size_t)255; return r; };
  f16* factors_h = (f16*)(ws + alloc((size_t)G_ * M_ * F_ * 2));
  f16* WqT  = (f16*)(ws + alloc((size_t)D_ * F_ * 2));
  f16* WkT  = (f16*)(ws + alloc((size_t)D_ * F_ * 2));
  f16* WvT  = (f16*)(ws + alloc((size_t)D_ * F_ * 2));
  f16* W1T  = (f16*)(ws + alloc((size_t)256 * 512 * 2));
  f16* W2T  = (f16*)(ws + alloc((size_t)128 * 256 * 2));
  f16* W3T  = (f16*)(ws + alloc((size_t)64 * 128 * 2));
  f16* q_h  = (f16*)(ws + alloc((size_t)G_ * M_ * D_ * 2));
  f16* k_h  = (f16*)(ws + alloc((size_t)G_ * M_ * D_ * 2));
  f16* vT   = (f16*)(ws + alloc((size_t)G_ * D_ * 32 * 2));
  float* w_ws = (float*)(ws + alloc((size_t)G_ * M_ * 4));

  // prep: convert / transpose weights into WMMA-friendly f16 layouts
  {
    int n = G_ * M_ * F_;
    k_cvt_f16<<<(n + 255) / 256, 256, 0, stream>>>(factors, factors_h, n);
  }
  k_transpose_f16<<<(F_ * D_ + 255) / 256, 256, 0, stream>>>(Wq, WqT, F_, D_);
  k_transpose_f16<<<(F_ * D_ + 255) / 256, 256, 0, stream>>>(Wk, WkT, F_, D_);
  k_transpose_f16<<<(F_ * D_ + 255) / 256, 256, 0, stream>>>(Wv, WvT, F_, D_);
  k_transpose_f16<<<(512 * 256 + 255) / 256, 256, 0, stream>>>(W1, W1T, 512, 256);
  k_transpose_f16<<<(256 * 128 + 255) / 256, 256, 0, stream>>>(W2, W2T, 256, 128);
  k_transpose_f16<<<(128 * 64 + 255) / 256, 256, 0, stream>>>(W3, W3T, 128, 64);

  // q/k/v projections: 64 groups x 3 mats x 32 D-tiles, one wave each
  k_qkv<<<G_ * 3 * 32, 32, 0, stream>>>(factors_h, WqT, WkT, WvT, q_h, k_h, vT);

  // attention + softmax + MLP -> w[g][m]
  k_att<<<G_, 32, 0, stream>>>(q_h, k_h, vT, lengths, W1T, b1, W2T, b2, W3T, b3,
                               W4, b4, w_ws);

  // memory-bound weighted reduction over raw (204.8 MB non-temporal stream)
  k_final<<<(N_ * G_) / 256, 256, 0, stream>>>(raw, w_ws, out);
}
